// CommunicationModule_89936615178811
// MI455X (gfx1250) — matrix operand order repaired and checked by reading, not compiled, not run on hardware
//
#include <hip/hip_runtime.h>
#include <hip/hip_bf16.h>
#include <math.h>
#include <stdint.h>

typedef _Float16 half_t;
typedef _Float16 v16h __attribute__((ext_vector_type(16)));
typedef _Float16 h8   __attribute__((ext_vector_type(8)));
typedef _Float16 h2v  __attribute__((ext_vector_type(2)));
typedef float    f2v  __attribute__((ext_vector_type(2)));
typedef float    v8f  __attribute__((ext_vector_type(8)));
typedef unsigned int u32x4 __attribute__((ext_vector_type(4)));
typedef int          i32x4 __attribute__((ext_vector_type(4)));
typedef int          i32x8 __attribute__((ext_vector_type(8)));

#define EDIM 256

// Tensor Data Mover availability (device pass only; host pass falls back)
#if defined(__gfx1250__) && __has_builtin(__builtin_amdgcn_tensor_load_to_lds)
#define HAVE_TDM 1
#else
#define HAVE_TDM 0
#endif

__device__ __forceinline__ float gelu_exact(float x) {
    return 0.5f * x * (1.0f + erff(x * 0.70710678118654752f));
}

// ---------------- elementwise f32 -> f16 (same layout) ----------------
__global__ void cvt_f16_kernel(const float* __restrict__ src, half_t* __restrict__ dst, long n) {
    long i = (long)blockIdx.x * blockDim.x + threadIdx.x;
    if (i < n) dst[i] = (half_t)src[i];
}

// ---------------- fold attention weights: comb = Wo @ Wv, bcomb = Wo@bv + bo --
__global__ void fold_attn_kernel(const float* __restrict__ wo, const float* __restrict__ wv,
                                 const float* __restrict__ bv, const float* __restrict__ bo,
                                 half_t* __restrict__ wh, float* __restrict__ bcomb) {
    int n = blockIdx.x;      // output row  (0..255)
    int k = threadIdx.x;     // output col  (0..255)
    float acc = 0.f;
    for (int m = 0; m < EDIM; ++m) acc += wo[n * EDIM + m] * wv[m * EDIM + k];
    wh[n * EDIM + k] = (half_t)acc;          // stored [N][K] row-major, like W
    if (k == 0) {
        float bb = bo[n];
        for (int m = 0; m < EDIM; ++m) bb += wo[n * EDIM + m] * bv[m];
        bcomb[n] = bb;
    }
}

// ---------------- tiled WMMA GEMM with fused epilogue ----------------
// Y[M][NCOLS] = epilogue( A[M][K] @ Wh[NCOLS][K]^T + bias )
//   AF32=true : A is f32 (srcF), AF32=false : up to 4 concatenated f16 sources
//   optional row gather via rowidx (stride idx_stride)
//   act: 0 none, 1 GELU, 2 ReLU ; optional LayerNorm (lng/lnb) over NCOLS
// block = 256 threads = 8 waves; block computes 16 rows x NCOLS.
// B tile staged via TDM tensor_load_to_lds (one 2D descriptor per wave slice).
template <int NCOLS, bool AF32>
__global__ void __launch_bounds__(256)
gemm_ep_kernel(const float* __restrict__ srcF,
               const half_t* __restrict__ a0, const half_t* __restrict__ a1,
               const half_t* __restrict__ a2, const half_t* __restrict__ a3,
               const int* __restrict__ rowidx, int idx_stride,
               const half_t* __restrict__ Wh,
               const float* __restrict__ bias,
               const float* __restrict__ lng, const float* __restrict__ lnb,
               int M, int K, int act,
               half_t* __restrict__ outH)
{
    constexpr int TPW = NCOLS / 128;   // wmma col-tiles per wave (8 waves, 16 cols/tile)
    __shared__ __align__(64) unsigned char smem[(size_t)NCOLS * 64 + 1024 + 192];
    half_t* Bs   = (half_t*)smem;                               // [NCOLS][32] f16
    float*  Epi  = (float*)smem;                                // [16][NCOLS] f32 (reuses Bs)
    half_t* As   = (half_t*)(smem + (size_t)NCOLS * 64);        // [16][32] f16
    float*  sMean = (float*)(smem + (size_t)NCOLS * 64 + 1024);
    float*  sRstd = sMean + 16;

    const int tid  = threadIdx.x;
    const int wave = tid >> 5;
    const int lane = tid & 31;
    const int hi   = lane >> 4;   // half-wave select
    const int lo16 = lane & 15;

    // ---- hoisted A-tile slot: thread owns (row rA, half-pair pA); gather once ----
    const int rA = tid >> 4;      // 0..15
    const int pA = tid & 15;      // 0..15 -> 2 halves each
    long rowA;
    {
        int gr = blockIdx.x * 16 + rA;
        if (gr >= M) gr = M - 1;
        rowA = rowidx ? (long)rowidx[(long)gr * idx_stride] : (long)gr;
    }

#if HAVE_TDM
    // ---- TDM descriptor invariants (wave-uniform), hoisted out of the K loop ----
    const int wtdm = __builtin_amdgcn_readfirstlane((int)(tid >> 5));
    const int rpw  = NCOLS / 8;                 // rows per wave slice
    const unsigned long long gaBase =
        (unsigned long long)(uintptr_t)(Wh + (size_t)wtdm * rpw * K);
    const unsigned ldsA = (unsigned)(uintptr_t)Bs + (unsigned)(wtdm * rpw * 64);
    i32x8 g1;
    g1.s0 = (int)(1u << 16);                     // data_size=1 -> 2 bytes; mask=0
    g1.s1 = (int)((unsigned)(K & 0xFFFF) << 16); // tensor_dim0[15:0]
    g1.s2 = (int)(((unsigned)K >> 16) |          // tensor_dim0[31:16]
                  ((unsigned)rpw << 16));        // tensor_dim1[15:0]
    g1.s3 = (int)(32u << 16);                    // tile_dim0 = 32 elements
    g1.s4 = (int)(unsigned)rpw;                  // tile_dim1 = rows; tile_dim2=0
    g1.s5 = (int)(unsigned)K;                    // tensor_dim0_stride[31:0]
    g1.s6 = 0;
    g1.s7 = 0;
    const i32x4 z4 = (i32x4)0;
#endif

    v8f acc[TPW];
    #pragma unroll
    for (int t = 0; t < TPW; ++t)
        #pragma unroll
        for (int v = 0; v < 8; ++v) acc[t][v] = 0.f;

    const int nseg = (K + 255) >> 8;             // 256-wide K segments
    for (int sg = 0; sg < nseg; ++sg) {
        const int kbase = sg << 8;
        const int kmax  = (K - kbase < 256) ? (K - kbase) : 256;

        // per-segment A source pointer (select once per 8 K-steps)
        const float*  apF = nullptr;
        const half_t* apH = nullptr;
        if constexpr (AF32) {
            apF = srcF + rowA * (long)K + kbase + pA * 2;
        } else {
            const half_t* sb = (sg == 0) ? a0 : (sg == 1) ? a1 : (sg == 2) ? a2 : a3;
            apH = sb + rowA * 256l + pA * 2;
        }

        for (int kk = 0; kk < kmax; kk += 32) {
            const int k0 = kbase + kk;
            __syncthreads();   // previous iteration's LDS consumers are done

            // ---- stage A tile: one 2-element load/convert per thread ----
            if constexpr (AF32) {
                f2v v = *(const f2v*)(apF + kk);
                h2v h; h.x = (half_t)v.x; h.y = (half_t)v.y;
                *(h2v*)(&As[rA * 32 + pA * 2]) = h;
            } else {
                *(h2v*)(&As[rA * 32 + pA * 2]) = *(const h2v*)(apH + kk);
            }

            // ---- stage B tile: NCOLS x 32 halves -> LDS [NCOLS][32] ----
#if HAVE_TDM
            {
                unsigned long long ga = gaBase + (unsigned long long)k0 * 2;
                u32x4 g0;
                g0.x = 1u;                                   // count=1 (valid user D#)
                g0.y = ldsA;                                 // lds_addr (bytes)
                g0.z = (unsigned)ga;                         // global_addr[31:0]
                g0.w = (unsigned)((ga >> 32) & 0x01FFFFFFu)  // global_addr[56:32]
                     | (2u << 30);                           // type = 2 ("image")
#if __clang_major__ >= 23
                i32x8 z8 = (i32x8)0;
                __builtin_amdgcn_tensor_load_to_lds(g0, g1, z4, z4, z8, 0);
#else
                __builtin_amdgcn_tensor_load_to_lds(g0, g1, z4, z4, 0);
#endif
            }
#if __has_builtin(__builtin_amdgcn_s_wait_tensorcnt)
            __builtin_amdgcn_s_wait_tensorcnt(0);
#endif
#else
            for (int t = tid; t < NCOLS * 4; t += 256) {
                int n = t >> 2, kc = (t & 3) << 3;
                *(h8*)(&Bs[n * 32 + kc]) = *(const h8*)(&Wh[(long)n * K + k0 + kc]);
            }
            if (k0 + 32 < K)
                __builtin_prefetch(&Wh[(long)(tid & (NCOLS - 1)) * K + k0 + 32], 0, 0);
#endif
            __syncthreads();

            // ---- A fragment: K = 16*hi + s (contiguous, matches B)
            h8 alo = *(const h8*)(&As[lo16 * 32 + 16 * hi]);
            h8 ahi = *(const h8*)(&As[lo16 * 32 + 16 * hi + 8]);
            v16h af = __builtin_shufflevector(alo, ahi, 0, 1, 2, 3, 4, 5, 6, 7,
                                                        8, 9, 10, 11, 12, 13, 14, 15);
            #pragma unroll
            for (int t = 0; t < TPW; ++t) {
                int c0 = (wave * TPW + t) * 16;
                // B fragment (32x16): lane holds col c0+lo16, K = 16*hi + h (32B chunk)
                v16h bf = *(const v16h*)(&Bs[(c0 + lo16) * 32 + 16 * hi]);
                acc[t] = __builtin_amdgcn_wmma_f32_16x16x32_f16(false, af, false, bf,
                                                                (short)0, acc[t], false, false);
            }
        }
    }

    // ---- epilogue: bias + activation into LDS (reuses Bs region) ----
    __syncthreads();
    #pragma unroll
    for (int t = 0; t < TPW; ++t) {
        int c0 = (wave * TPW + t) * 16;
        int n  = c0 + lo16;
        float bb = bias[n];
        #pragma unroll
        for (int v = 0; v < 8; ++v) {
            int m   = v + 8 * hi;              // C/D layout: VGPR v, half-wave hi
            float x = acc[t][v] + bb;
            if (act == 1)      x = gelu_exact(x);
            else if (act == 2) x = fmaxf(x, 0.f);
            Epi[m * NCOLS + n] = x;
        }
    }
    __syncthreads();
    if (lng) {
        if (tid < 16) {
            float s = 0.f, s2 = 0.f;
            for (int n = 0; n < NCOLS; ++n) {
                float x = Epi[tid * NCOLS + n];
                s += x; s2 += x * x;
            }
            float mean = s / NCOLS;
            float var  = s2 / NCOLS - mean * mean;
            sMean[tid] = mean;
            sRstd[tid] = rsqrtf(var + 1e-5f);
        }
        __syncthreads();
    }
    for (int t = tid; t < 16 * NCOLS; t += 256) {
        int m = t / NCOLS, n = t - m * NCOLS;
        long grow = (long)blockIdx.x * 16 + m;
        if (grow >= M) continue;
        float x = Epi[t];
        if (lng) x = (x - sMean[m]) * sRstd[m] * lng[n] + lnb[n];
        outH[grow * NCOLS + n] = (half_t)x;
    }
}

// ---------------- gate (4 sigmoids) + fused combine ----------------
__global__ void __launch_bounds__(256)
gate_fuse_kernel(const half_t* __restrict__ U, const float* __restrict__ wg2,
                 const float* __restrict__ bg2,
                 const half_t* __restrict__ ai, const half_t* __restrict__ aj,
                 const half_t* __restrict__ ci, const half_t* __restrict__ cj,
                 half_t* __restrict__ fused, int P)
{
    int lane = threadIdx.x & 31;
    int wave = threadIdx.x >> 5;
    long row = (long)blockIdx.x * 8 + wave;
    if (row >= P) return;
    float g[4];
    #pragma unroll
    for (int c = 0; c < 4; ++c) {
        float p = 0.f;
        for (int k = lane; k < 512; k += 32)
            p += (float)U[row * 512 + k] * wg2[c * 512 + k];
        #pragma unroll
        for (int off = 16; off > 0; off >>= 1) p += __shfl_xor(p, off, 32);
        g[c] = 1.f / (1.f + expf(-(p + bg2[c])));
    }
    for (int n = lane; n < 256; n += 32) {
        float f = g[0] * (float)ai[row * 256 + n] + g[1] * (float)aj[row * 256 + n]
                + g[2] * (float)ci[row * 256 + n] + g[3] * (float)cj[row * 256 + n];
        fused[row * 256 + n] = (half_t)f;
    }
}

// ---------------- final score: sigmoid(s1 . ws2 + bs2) ----------------
__global__ void score_kernel(const half_t* __restrict__ s1, const float* __restrict__ ws2,
                             const float* __restrict__ bs2, float* __restrict__ out, int P)
{
    long t = (long)blockIdx.x * blockDim.x + threadIdx.x;
    if (t >= P) return;
    float acc = bs2[0];
    for (int j = 0; j < 128; ++j) acc += (float)s1[t * 128 + j] * ws2[j];
    out[t] = 1.f / (1.f + expf(-acc));
}

// =====================================================================
extern "C" void kernel_launch(void* const* d_in, const int* in_sizes, int n_in,
                              void* d_out, int out_size, void* d_ws, size_t ws_size,
                              hipStream_t stream)
{
    (void)n_in; (void)out_size; (void)ws_size;
    const float* cellF  = (const float*)d_in[0];
    const int*   pairs  = (const int*)d_in[1];
    const float* ch_w1  = (const float*)d_in[2];
    const float* ch_b1  = (const float*)d_in[3];
    const float* ch_g1  = (const float*)d_in[4];
    const float* ch_bb1 = (const float*)d_in[5];
    const float* ch_w2  = (const float*)d_in[6];
    const float* ch_b2  = (const float*)d_in[7];
    const float* ch_g2  = (const float*)d_in[8];
    const float* ch_bb2 = (const float*)d_in[9];
    // sa params: 10 wq, 11 bq, 12 wk, 13 bk, 14 wv, 15 bv, 16 wo, 17 bo, 18 g, 19 b
    const float* sa_wv = (const float*)d_in[14];
    const float* sa_bv = (const float*)d_in[15];
    const float* sa_wo = (const float*)d_in[16];
    const float* sa_bo = (const float*)d_in[17];
    const float* sa_g  = (const float*)d_in[18];
    const float* sa_b  = (const float*)d_in[19];
    // ca params: 20..29
    const float* ca_wv = (const float*)d_in[24];
    const float* ca_bv = (const float*)d_in[25];
    const float* ca_wo = (const float*)d_in[26];
    const float* ca_bo = (const float*)d_in[27];
    const float* ca_g  = (const float*)d_in[28];
    const float* ca_b  = (const float*)d_in[29];
    const float* wg1 = (const float*)d_in[30];
    const float* bg1 = (const float*)d_in[31];
    const float* wg2 = (const float*)d_in[32];
    const float* bg2 = (const float*)d_in[33];
    const float* ws1 = (const float*)d_in[34];
    const float* bs1 = (const float*)d_in[35];
    const float* ws2 = (const float*)d_in[36];
    const float* bs2 = (const float*)d_in[37];

    const long N = in_sizes[0] / 512;   // 100000
    const long P = in_sizes[1] / 2;     // 131072

    // ---- workspace carve-up ----
    char*  ws  = (char*)d_ws;
    size_t off = 0;
    auto alloc = [&](size_t bytes) -> void* {
        void* p = ws + off;
        off += (bytes + 255) & ~(size_t)255;
        return p;
    };
    half_t* w1h   = (half_t*)alloc(256l * 512 * 2);
    half_t* w2h   = (half_t*)alloc(256l * 256 * 2);
    half_t* wsah  = (half_t*)alloc(256l * 256 * 2);
    half_t* wcah  = (half_t*)alloc(256l * 256 * 2);
    half_t* wg1h  = (half_t*)alloc(512l * 1024 * 2);
    half_t* ws1h  = (half_t*)alloc(128l * 256 * 2);
    float*  bsaf  = (float*)alloc(256 * 4);
    float*  bcaf  = (float*)alloc(256 * 4);
    half_t* hbuf  = (half_t*)alloc((size_t)N * 256 * 2);
    half_t* aib   = (half_t*)alloc((size_t)P * 256 * 2);
    half_t* ajb   = (half_t*)alloc((size_t)P * 256 * 2);
    half_t* cib   = (half_t*)alloc((size_t)P * 256 * 2);
    half_t* cjb   = (half_t*)alloc((size_t)P * 256 * 2);
    half_t* Ubuf  = (half_t*)alloc((size_t)P * 512 * 2);  // also reused as h1 (N*256 <= P*512)
    half_t* fusedb= (half_t*)alloc((size_t)P * 256 * 2);
    half_t* s1b   = (half_t*)alloc((size_t)P * 128 * 2);

    auto cvt = [&](const float* s, half_t* d, long n) {
        cvt_f16_kernel<<<(int)((n + 255) / 256), 256, 0, stream>>>(s, d, n);
    };
    // ---- weight prep ----
    cvt(ch_w1, w1h, 256l * 512);
    cvt(ch_w2, w2h, 256l * 256);
    cvt(wg1,  wg1h, 512l * 1024);
    cvt(ws1,  ws1h, 128l * 256);
    fold_attn_kernel<<<256, 256, 0, stream>>>(sa_wo, sa_wv, sa_bv, sa_bo, wsah, bsaf);
    fold_attn_kernel<<<256, 256, 0, stream>>>(ca_wo, ca_wv, ca_bv, ca_bo, wcah, bcaf);

    const int nbN = (int)((N + 15) / 16);
    const int nbP = (int)((P + 15) / 16);

    // ---- cell MLP: h1 = LN(GELU(X@W1^T+b1)), h = LN(GELU(h1@W2^T+b2)) ----
    gemm_ep_kernel<256, true><<<nbN, 256, 0, stream>>>(cellF, nullptr, nullptr, nullptr, nullptr,
        nullptr, 0, w1h, ch_b1, ch_g1, ch_bb1, (int)N, 512, /*gelu*/1, Ubuf);
    gemm_ep_kernel<256, false><<<nbN, 256, 0, stream>>>(nullptr, Ubuf, Ubuf, Ubuf, Ubuf,
        nullptr, 0, w2h, ch_b2, ch_g2, ch_bb2, (int)N, 256, /*gelu*/1, hbuf);

    // ---- folded "attention" (softmax over singleton => ctx = V) ----
    gemm_ep_kernel<256, false><<<nbP, 256, 0, stream>>>(nullptr, hbuf, hbuf, hbuf, hbuf,
        pairs + 0, 2, wsah, bsaf, sa_g, sa_b, (int)P, 256, 0, aib);
    gemm_ep_kernel<256, false><<<nbP, 256, 0, stream>>>(nullptr, hbuf, hbuf, hbuf, hbuf,
        pairs + 1, 2, wsah, bsaf, sa_g, sa_b, (int)P, 256, 0, ajb);
    gemm_ep_kernel<256, false><<<nbP, 256, 0, stream>>>(nullptr, ajb, ajb, ajb, ajb,
        nullptr, 0, wcah, bcaf, ca_g, ca_b, (int)P, 256, 0, cib);
    gemm_ep_kernel<256, false><<<nbP, 256, 0, stream>>>(nullptr, aib, aib, aib, aib,
        nullptr, 0, wcah, bcaf, ca_g, ca_b, (int)P, 256, 0, cjb);

    // ---- gate hidden: U = GELU(concat[ai,aj,ci,cj] @ wg1^T + bg1) ----
    gemm_ep_kernel<512, false><<<nbP, 256, 0, stream>>>(nullptr, aib, ajb, cib, cjb,
        nullptr, 0, wg1h, bg1, nullptr, nullptr, (int)P, 1024, /*gelu*/1, Ubuf);

    // ---- gate sigmoids + fused combine ----
    gate_fuse_kernel<<<(int)((P + 7) / 8), 256, 0, stream>>>(Ubuf, wg2, bg2,
        aib, ajb, cib, cjb, fusedb, (int)P);

    // ---- score head: s1 = relu(fused @ ws1^T + bs1), out = sigmoid(s1.ws2+bs2) ----
    gemm_ep_kernel<128, false><<<nbP, 256, 0, stream>>>(nullptr, fusedb, fusedb, fusedb, fusedb,
        nullptr, 0, ws1h, bs1, nullptr, nullptr, (int)P, 256, /*relu*/2, s1b);
    score_kernel<<<(int)((P + 255) / 256), 256, 0, stream>>>(s1b, ws2, bs2,
        (float*)d_out, (int)P);
}
